// PrunedSelfAttentionBlock_30597347017100
// MI455X (gfx1250) — matrix operand order repaired and checked by reading, not compiled, not run on hardware
//
#include <hip/hip_runtime.h>
#include <hip/hip_bf16.h>

// ---------------------------------------------------------------------------
// Problem constants (shapes fixed by the reference)
// ---------------------------------------------------------------------------
#define BATCH 2
#define CDIM 512
#define NTOK 2304      // H*W = 48*48
#define MLPD 2048
#define NHEADS 8
#define DHEAD 64
#define TOPK 128
#define BN_EPS 1e-5f

typedef __bf16 bf16;
typedef __attribute__((ext_vector_type(16))) __bf16 v16bf;
typedef __attribute__((ext_vector_type(8)))  float  v8f;
typedef __attribute__((ext_vector_type(4)))  unsigned su4;
typedef __attribute__((ext_vector_type(8)))  unsigned su8;

struct alignas(16) U128 { unsigned int x[4]; };

static __device__ inline v16bf mkfrag(U128 lo, U128 hi) {
    union { U128 u[2]; v16bf v; } f;
    f.u[0] = lo; f.u[1] = hi;
    return f.v;
}
static __device__ inline U128 ld128(const bf16* p) { return *(const U128*)p; }

static __device__ inline v8f wmma_bf16(v16bf a, v16bf b, v8f c) {
    return __builtin_amdgcn_wmma_f32_16x16x32_bf16(
        /*neg_a=*/false, a, /*neg_b=*/false, b,
        /*c_mod=*/(short)0, c, /*reuse_a=*/false, /*reuse_b=*/false);
}

static __device__ inline float wred_max(float v) {
    #pragma unroll
    for (int o = 16; o > 0; o >>= 1) v = fmaxf(v, __shfl_xor(v, o, 32));
    return v;
}
static __device__ inline float wred_sum(float v) {
    #pragma unroll
    for (int o = 16; o > 0; o >>= 1) v += __shfl_xor(v, o, 32);
    return v;
}
static __device__ inline int wred_sumi(int v) {
    #pragma unroll
    for (int o = 16; o > 0; o >>= 1) v += __shfl_xor(v, o, 32);
    return v;
}

// ---------------------------------------------------------------------------
// TDM: issue one 2D tile load Bt[row0..row0+63][k0..k0+31] (bf16) -> LDS.
// LDS layout after pad: row stride = 32 elems * 2B + 16B pad = 80 B (= [64][40]).
// All inputs must be wave-uniform (SGPR-constrained inline asm).
// ---------------------------------------------------------------------------
static __device__ inline void tdm_load_tile_b64x32(const bf16* tileBase, int k0,
                                                   unsigned ldsOff, int K) {
    const unsigned long long ga =
        (unsigned long long)(__UINTPTR_TYPE__)(tileBase + k0);
    su4 g0;
    g0[0] = 1u;                                   // count=1, user mode
    g0[1] = ldsOff;                               // LDS byte address
    g0[2] = (unsigned)(ga & 0xffffffffu);         // global_addr[31:0]
    g0[3] = (unsigned)((ga >> 32) & 0x01ffffffu)  // global_addr[56:32]
          | (2u << 30);                           // type=2 (image)
    su8 g1;
    g1[0] = (1u << 16)     // data_size = 1 -> 2 bytes
          | (1u << 20)     // pad_enable
          | (3u << 22)     // pad_interval = 16 DWORDs (one 32-elem bf16 row)
          | (3u << 25);    // pad_amount   = 4 DWORDs (16 B)
    g1[1] = ((unsigned)K & 0xffffu) << 16;        // tensor_dim0[15:0]
    g1[2] = (((unsigned)K >> 16) & 0xffffu)       // tensor_dim0[31:16]
          | (64u << 16);                          // tensor_dim1[15:0] = 64 rows
    g1[3] = (32u << 16);                          // tile_dim0 = 32 (td1 hi = 0)
    g1[4] = 64u;                                  // tile_dim1 = 64, tile_dim2 = 0
    g1[5] = (unsigned)K;                          // tensor_dim0_stride[31:0]
    g1[6] = 0u;                                   // stride hi, dim1_stride lo
    g1[7] = 0u;
    asm volatile("tensor_load_to_lds %0, %1" :: "s"(g0), "s"(g1) : "memory");
}

// ---------------------------------------------------------------------------
// Generic tiled bf16 WMMA GEMM: out = A[M,K] * Bt[N,K]^T (+bias) (+residual)
// Block: 256 thr = 8 waves. Block tile: 128(M) x 64(N). Wave tile: 16 x 64.
// B tiles double-buffered in LDS via TDM (wave 0 drives the DMA).
// Grid: (N/64, M/128, BATCH).
// ---------------------------------------------------------------------------
enum { EP_BF16_T = 0,   // store bf16 transposed [N,M] (ld = M)
       EP_BF16_N = 1,   // store bf16 normal     [M,N]
       EP_F32    = 2,   // store fp32 normal     [M,N]
       EP_F32_RES= 3 }; // store fp32 normal + residual

template<int EPI>
__global__ __launch_bounds__(256)
void gemm_bf16_kernel(const bf16* __restrict__ A,     // [M,K]
                      const bf16* __restrict__ Bt0,   // [BATCH][N,K]
                      const float* __restrict__ bias, // [M] (may be null)
                      const float* __restrict__ res0, // [BATCH][M,N] (mode 3)
                      void* __restrict__ out0,        // per-batch output
                      int M, int N, int K)
{
    const int z     = blockIdx.z;
    const int mBase = blockIdx.y * 128;
    const int nBase = blockIdx.x * 64;
    const int tid   = threadIdx.x;
    const int lane  = tid & 31;
    const int wv    = tid >> 5;
    const int l16   = lane & 15;
    const int half  = lane >> 4;

    __shared__ __attribute__((aligned(16))) bf16 sB[2][64][40]; // 2 x (64 rows x 32 K + pad)

    const bf16* tileBase = Bt0 + (size_t)z * N * K + (size_t)nBase * K;
    const unsigned ldsOff[2] = {
        (unsigned)(__UINTPTR_TYPE__)&sB[0][0][0],
        (unsigned)(__UINTPTR_TYPE__)&sB[1][0][0]
    };

    const int mRow = mBase + wv * 16 + l16;      // A row owned by this lane
    const bf16* Arow = A + (size_t)mRow * K;

    v8f acc0 = {}, acc1 = {}, acc2 = {}, acc3 = {};

    if (tid < 32)    // wave 0 drives the tensor DMA
        tdm_load_tile_b64x32(tileBase, 0, ldsOff[0], K);

    for (int k0 = 0; k0 < K; k0 += 32) {
        const int buf = (k0 >> 5) & 1;
        if (tid < 32) {
            if (k0 + 32 < K) {
                tdm_load_tile_b64x32(tileBase, k0 + 32, ldsOff[buf ^ 1], K);
                __builtin_amdgcn_s_wait_tensorcnt(1);  // current tile landed
            } else {
                __builtin_amdgcn_s_wait_tensorcnt(0);
            }
        }
        __syncthreads();

        // A fragment: runs K=[k0+8h, +8) and [k0+16+8h, +8)
        v16bf af = mkfrag(ld128(Arow + k0 + 8 * half),
                          ld128(Arow + k0 + 16 + 8 * half));

        // 4 B fragments from LDS, col = t*16 + l16, runs K=[8h, +8) and [16+8h, +8)
        v16bf b0 = mkfrag(*(const U128*)&sB[buf][ 0 + l16][8 * half],
                          *(const U128*)&sB[buf][ 0 + l16][16 + 8 * half]);
        v16bf b1 = mkfrag(*(const U128*)&sB[buf][16 + l16][8 * half],
                          *(const U128*)&sB[buf][16 + l16][16 + 8 * half]);
        v16bf b2 = mkfrag(*(const U128*)&sB[buf][32 + l16][8 * half],
                          *(const U128*)&sB[buf][32 + l16][16 + 8 * half]);
        v16bf b3 = mkfrag(*(const U128*)&sB[buf][48 + l16][8 * half],
                          *(const U128*)&sB[buf][48 + l16][16 + 8 * half]);

        acc0 = wmma_bf16(af, b0, acc0);
        acc1 = wmma_bf16(af, b1, acc1);
        acc2 = wmma_bf16(af, b2, acc2);
        acc3 = wmma_bf16(af, b3, acc3);

        __syncthreads();   // tile consumed; DMA may overwrite it next step
    }

    // epilogue
    #pragma unroll
    for (int t = 0; t < 4; ++t) {
        const v8f acc = (t == 0) ? acc0 : (t == 1) ? acc1 : (t == 2) ? acc2 : acc3;
        const int n = nBase + t * 16 + l16;
        #pragma unroll
        for (int r = 0; r < 8; ++r) {
            const int m = mBase + wv * 16 + r + 8 * half;
            float v = acc[r];
            if (bias) v += bias[m];
            if (EPI == EP_BF16_T) {
                ((bf16*)out0)[(size_t)z * N * M + (size_t)n * M + m] = (bf16)v;
            } else if (EPI == EP_BF16_N) {
                ((bf16*)out0)[(size_t)z * M * N + (size_t)m * N + n] = (bf16)v;
            } else if (EPI == EP_F32) {
                ((float*)out0)[(size_t)z * M * N + (size_t)m * N + n] = v;
            } else {
                const size_t ix = (size_t)z * M * N + (size_t)m * N + n;
                ((float*)out0)[ix] = v + res0[ix];
            }
        }
    }
}

// ---------------------------------------------------------------------------
// Fused pruned attention: one block (4 waves) per (16-query tile, head, batch)
// qt,kt: [B][N][C] bf16 (token-major). v: [B][C][N] bf16 (channel-major).
// attnt out: [B][N][C] bf16.  Dynamic LDS: scores fp32 + attn bf16.
// ---------------------------------------------------------------------------
#define S_STRIDE 2308   // NTOK + 4 floats
#define A_STRIDE 2320   // NTOK + 16 halfs (keeps 16B alignment)
#define ATTN_SMEM_BYTES (16 * S_STRIDE * 4 + 16 * A_STRIDE * 2)

__global__ __launch_bounds__(128)
void attn_topk_kernel(const bf16* __restrict__ qt,
                      const bf16* __restrict__ kt,
                      const bf16* __restrict__ vc,
                      bf16* __restrict__ attnt)
{
    extern __shared__ char smem[];
    float* S  = (float*)smem;                          // [16][S_STRIDE]
    bf16*  AW = (bf16*)(smem + 16 * S_STRIDE * 4);     // [16][A_STRIDE]

    const int mBase = blockIdx.x * 16;        // query tile
    const int h     = blockIdx.y & (NHEADS - 1);
    const int b     = blockIdx.y >> 3;
    const int lane  = threadIdx.x & 31;
    const int wv    = threadIdx.x >> 5;       // 4 waves
    const int l16   = lane & 15;
    const int half  = lane >> 4;

    const bf16* qb = qt + ((size_t)b * NTOK) * CDIM + (size_t)h * DHEAD;
    const bf16* kb = kt + ((size_t)b * NTOK) * CDIM + (size_t)h * DHEAD;
    const bf16* vb = vc + ((size_t)b * CDIM + (size_t)h * DHEAD) * NTOK;
    bf16*       ab = attnt + ((size_t)b * NTOK) * CDIM + (size_t)h * DHEAD;

    // ---- Phase 1: scores tile S[16][NTOK] = (Q * K^T) * rsqrt(d) -----------
    {
        const bf16* qp = qb + (size_t)(mBase + l16) * CDIM;
        v16bf a0 = mkfrag(ld128(qp +       8 * half), ld128(qp + 16 + 8 * half));
        v16bf a1 = mkfrag(ld128(qp + 32 +  8 * half), ld128(qp + 48 + 8 * half));
        for (int mt = wv; mt < NTOK / 16; mt += 4) {
            const bf16* kp = kb + (size_t)(mt * 16 + l16) * CDIM;
            v8f c = {};
            c = wmma_bf16(a0, mkfrag(ld128(kp + 8 * half),
                                     ld128(kp + 16 + 8 * half)), c);
            c = wmma_bf16(a1, mkfrag(ld128(kp + 32 + 8 * half),
                                     ld128(kp + 48 + 8 * half)), c);
            #pragma unroll
            for (int r = 0; r < 8; ++r)
                S[(r + 8 * half) * S_STRIDE + mt * 16 + l16] = c[r] * 0.125f;
        }
    }
    __syncthreads();

    // ---- Phase 2/3: per-row exact top-k threshold + masked softmax ---------
    for (int row = wv; row < 16; row += 4) {
        float* Sr = S + row * S_STRIDE;
        bf16*  Ar = AW + row * A_STRIDE;

        float mx = -3.4e38f;
        for (int m = lane; m < NTOK; m += 32) mx = fmaxf(mx, Sr[m]);
        mx = wred_max(mx);

        // exact kth-largest via 32-step bisection on order-preserving keys
        unsigned thr = 0u;
        for (int bit = 31; bit >= 0; --bit) {
            const unsigned cand = thr | (1u << bit);
            int cnt = 0;
            for (int m = lane; m < NTOK; m += 32) {
                unsigned u = __float_as_uint(Sr[m]);
                unsigned key = (u & 0x80000000u) ? ~u : (u | 0x80000000u);
                cnt += (key >= cand) ? 1 : 0;
            }
            cnt = wred_sumi(cnt);
            if (cnt >= TOPK) thr = cand;
        }

        float s = 0.f;
        for (int m = lane; m < NTOK; m += 32) {
            unsigned u = __float_as_uint(Sr[m]);
            unsigned key = (u & 0x80000000u) ? ~u : (u | 0x80000000u);
            if (key >= thr) s += __expf(Sr[m] - mx);
        }
        s = wred_sum(s);
        const float inv = 1.0f / s;

        for (int m = lane; m < NTOK; m += 32) {
            unsigned u = __float_as_uint(Sr[m]);
            unsigned key = (u & 0x80000000u) ? ~u : (u | 0x80000000u);
            Ar[m] = (key >= thr) ? (bf16)(__expf(Sr[m] - mx) * inv) : (bf16)0.0f;
        }
    }
    __syncthreads();

    // ---- Phase 4: O tile = attn[16,NTOK] * V[NTOK,16] per wave d-tile ------
    {
        const int dt = wv;                       // 4 waves x 16 cols = 64 = DHEAD
        const bf16* Ar = AW + (size_t)l16 * A_STRIDE;
        const bf16* vr = vb + (size_t)(dt * 16 + l16) * NTOK;
        v8f c = {};
        for (int k0 = 0; k0 < NTOK; k0 += 32) {
            v16bf af = mkfrag(*(const U128*)(Ar + k0 + 8 * half),
                              *(const U128*)(Ar + k0 + 16 + 8 * half));
            v16bf bf_ = mkfrag(ld128(vr + k0 + 8 * half),
                               ld128(vr + k0 + 16 + 8 * half));
            c = wmma_bf16(af, bf_, c);
        }
        #pragma unroll
        for (int r = 0; r < 8; ++r) {
            const int n = mBase + r + 8 * half;
            ab[(size_t)n * CDIM + dt * 16 + l16] = (bf16)c[r];
        }
    }
}

// ---------------------------------------------------------------------------
// Helpers: casts / transposes / batchnorm
// ---------------------------------------------------------------------------
__global__ __launch_bounds__(256)
void cast_bf16_kernel(const float* __restrict__ s, bf16* __restrict__ d, int n) {
    int i = blockIdx.x * 256 + threadIdx.x;
    if (i < n) d[i] = (bf16)s[i];
}

// fp32 [B][CH][N] -> bf16 [B][N][CH]
__global__ __launch_bounds__(256)
void transpose_cast_kernel(const float* __restrict__ s, bf16* __restrict__ d,
                           int CH, int N) {
    const int z = blockIdx.z;
    const int i = blockIdx.x * 256 + threadIdx.x;
    if (i >= CH * N) return;
    const int ch = i / N, n = i % N;
    d[((size_t)z * N + n) * CH + ch] = (bf16)s[((size_t)z * CH + ch) * N + n];
}

// per-channel mean/rstd over (B, N), biased variance
__global__ __launch_bounds__(256)
void bn_stats_kernel(const float* __restrict__ y, float* __restrict__ mean,
                     float* __restrict__ rstd, int CH, int N) {
    const int ch = blockIdx.x;
    float s = 0.f, ss = 0.f;
    for (int i = threadIdx.x; i < BATCH * N; i += 256) {
        const int b = i / N, n = i % N;
        const float v = y[((size_t)b * CH + ch) * N + n];
        s += v; ss += v * v;
    }
    __shared__ float sm[256], sq[256];
    sm[threadIdx.x] = s; sq[threadIdx.x] = ss;
    __syncthreads();
    for (int w = 128; w > 0; w >>= 1) {
        if (threadIdx.x < w) {
            sm[threadIdx.x] += sm[threadIdx.x + w];
            sq[threadIdx.x] += sq[threadIdx.x + w];
        }
        __syncthreads();
    }
    if (threadIdx.x == 0) {
        const float inv = 1.0f / (float)(BATCH * N);
        const float m = sm[0] * inv;
        const float var = sq[0] * inv - m * m;
        mean[ch] = m;
        rstd[ch] = rsqrtf(var + BN_EPS);
    }
}

// BN + ReLU + transpose-cast: fp32 [B][CH][N] -> bf16 [B][N][CH]
__global__ __launch_bounds__(256)
void bn_relu_t_kernel(const float* __restrict__ y, const float* __restrict__ mean,
                      const float* __restrict__ rstd, const float* __restrict__ g,
                      const float* __restrict__ be, bf16* __restrict__ dT,
                      int CH, int N) {
    const int z = blockIdx.z;
    const int i = blockIdx.x * 256 + threadIdx.x;
    if (i >= CH * N) return;
    const int ch = i / N, n = i % N;
    float v = y[((size_t)z * CH + ch) * N + n];
    v = (v - mean[ch]) * rstd[ch] * g[ch] + be[ch];
    v = fmaxf(v, 0.f);
    dT[((size_t)z * N + n) * CH + ch] = (bf16)v;
}

// out = BN2(h2) + o   (fp32, channel-major)
__global__ __launch_bounds__(256)
void bn_residual_kernel(const float* __restrict__ y, const float* __restrict__ mean,
                        const float* __restrict__ rstd, const float* __restrict__ g,
                        const float* __restrict__ be, const float* __restrict__ o,
                        float* __restrict__ out, int CH, int N) {
    const int z = blockIdx.z;
    const int i = blockIdx.x * 256 + threadIdx.x;
    if (i >= CH * N) return;
    const int ch = i / N;
    const size_t ix = (size_t)z * CH * N + i;
    const float v = (y[ix] - mean[ch]) * rstd[ch] * g[ch] + be[ch];
    out[ix] = v + o[ix];
}

// ---------------------------------------------------------------------------
// Orchestration
// ---------------------------------------------------------------------------
extern "C" void kernel_launch(void* const* d_in, const int* in_sizes, int n_in,
                              void* d_out, int out_size, void* d_ws, size_t ws_size,
                              hipStream_t stream) {
    (void)in_sizes; (void)n_in; (void)out_size; (void)ws_size;

    const float* x  = (const float*)d_in[0];
    const float* Wq = (const float*)d_in[1];  const float* bq = (const float*)d_in[2];
    const float* Wk = (const float*)d_in[3];  const float* bk = (const float*)d_in[4];
    const float* Wv = (const float*)d_in[5];  const float* bv = (const float*)d_in[6];
    const float* Wo = (const float*)d_in[7];  const float* bo = (const float*)d_in[8];
    const float* W1 = (const float*)d_in[9];  const float* b1 = (const float*)d_in[10];
    const float* g1 = (const float*)d_in[11]; const float* be1= (const float*)d_in[12];
    const float* W2 = (const float*)d_in[13]; const float* b2 = (const float*)d_in[14];
    const float* g2 = (const float*)d_in[15]; const float* be2= (const float*)d_in[16];
    float* out = (float*)d_out;

    // workspace carve-out
    size_t off = 0;
    auto take = [&](size_t bytes) {
        void* p = (char*)d_ws + off;
        off += (bytes + 255) & ~(size_t)255;
        return p;
    };
    bf16* wqb  = (bf16*)take((size_t)CDIM * CDIM * 2);
    bf16* wkb  = (bf16*)take((size_t)CDIM * CDIM * 2);
    bf16* wvb  = (bf16*)take((size_t)CDIM * CDIM * 2);
    bf16* wob  = (bf16*)take((size_t)CDIM * CDIM * 2);
    bf16* w1b  = (bf16*)take((size_t)MLPD * CDIM * 2);
    bf16* w2b  = (bf16*)take((size_t)CDIM * MLPD * 2);
    bf16* xt   = (bf16*)take((size_t)BATCH * NTOK * CDIM * 2);
    bf16* qt   = (bf16*)take((size_t)BATCH * NTOK * CDIM * 2);
    bf16* kt   = (bf16*)take((size_t)BATCH * NTOK * CDIM * 2);
    bf16* vcb  = (bf16*)take((size_t)BATCH * CDIM * NTOK * 2);
    bf16* attnt= (bf16*)take((size_t)BATCH * NTOK * CDIM * 2);
    float* obuf= (float*)take((size_t)BATCH * CDIM * NTOK * 4);
    bf16* ot   = (bf16*)take((size_t)BATCH * NTOK * CDIM * 2);
    float* h1  = (float*)take((size_t)BATCH * MLPD * NTOK * 4);
    bf16* h1t  = (bf16*)take((size_t)BATCH * NTOK * MLPD * 2);
    float* h2  = (float*)take((size_t)BATCH * CDIM * NTOK * 4);
    float* mean1 = (float*)take((size_t)MLPD * 4);
    float* rstd1 = (float*)take((size_t)MLPD * 4);
    float* mean2 = (float*)take((size_t)CDIM * 4);
    float* rstd2 = (float*)take((size_t)CDIM * 4);

    // 1) weight casts fp32 -> bf16
    const int wc = CDIM * CDIM, w1n = MLPD * CDIM;
    cast_bf16_kernel<<<(wc + 255) / 256, 256, 0, stream>>>(Wq, wqb, wc);
    cast_bf16_kernel<<<(wc + 255) / 256, 256, 0, stream>>>(Wk, wkb, wc);
    cast_bf16_kernel<<<(wc + 255) / 256, 256, 0, stream>>>(Wv, wvb, wc);
    cast_bf16_kernel<<<(wc + 255) / 256, 256, 0, stream>>>(Wo, wob, wc);
    cast_bf16_kernel<<<(w1n + 255) / 256, 256, 0, stream>>>(W1, w1b, w1n);
    cast_bf16_kernel<<<(w1n + 255) / 256, 256, 0, stream>>>(W2, w2b, w1n);

    const int cn = CDIM * NTOK;
    dim3 gEl((cn + 255) / 256, 1, BATCH);

    // 2) x -> xt  (token-major bf16)
    transpose_cast_kernel<<<gEl, 256, 0, stream>>>(x, xt, CDIM, NTOK);

    // 3) Q/K/V projections (GEMM, K=C)
    dim3 gP(NTOK / 64, CDIM / 128, BATCH);
    gemm_bf16_kernel<EP_BF16_T><<<gP, 256, 0, stream>>>(wqb, xt, bq, nullptr, qt,
                                                        CDIM, NTOK, CDIM);
    gemm_bf16_kernel<EP_BF16_T><<<gP, 256, 0, stream>>>(wkb, xt, bk, nullptr, kt,
                                                        CDIM, NTOK, CDIM);
    gemm_bf16_kernel<EP_BF16_N><<<gP, 256, 0, stream>>>(wvb, xt, bv, nullptr, vcb,
                                                        CDIM, NTOK, CDIM);

    // 4) fused top-k attention
    dim3 gA(NTOK / 16, NHEADS * BATCH, 1);
    attn_topk_kernel<<<gA, 128, ATTN_SMEM_BYTES, stream>>>(qt, kt, vcb, attnt);

    // 5) out projection + residual x -> obuf (fp32, channel-major)
    gemm_bf16_kernel<EP_F32_RES><<<gP, 256, 0, stream>>>(wob, attnt, bo, x, obuf,
                                                         CDIM, NTOK, CDIM);

    // 6) obuf -> ot (token-major bf16) for MLP GEMM1
    transpose_cast_kernel<<<gEl, 256, 0, stream>>>(obuf, ot, CDIM, NTOK);

    // 7) MLP GEMM1: h1 = W1 * o + b1   (M=MLPD, K=CDIM)
    dim3 g1d(NTOK / 64, MLPD / 128, BATCH);
    gemm_bf16_kernel<EP_F32><<<g1d, 256, 0, stream>>>(w1b, ot, b1, nullptr, h1,
                                                      MLPD, NTOK, CDIM);

    // 8) BN1 stats + apply(ReLU) + transpose-cast
    bn_stats_kernel<<<MLPD, 256, 0, stream>>>(h1, mean1, rstd1, MLPD, NTOK);
    dim3 gB1((MLPD * NTOK + 255) / 256, 1, BATCH);
    bn_relu_t_kernel<<<gB1, 256, 0, stream>>>(h1, mean1, rstd1, g1, be1, h1t,
                                              MLPD, NTOK);

    // 9) MLP GEMM2: h2 = W2 * h1relu + b2  (M=CDIM, K=MLPD)
    gemm_bf16_kernel<EP_F32><<<gP, 256, 0, stream>>>(w2b, h1t, b2, nullptr, h2,
                                                     CDIM, NTOK, MLPD);

    // 10) BN2 stats + BN + residual -> out
    bn_stats_kernel<<<CDIM, 256, 0, stream>>>(h2, mean2, rstd2, CDIM, NTOK);
    bn_residual_kernel<<<gEl, 256, 0, stream>>>(h2, mean2, rstd2, g2, be2, obuf,
                                                out, CDIM, NTOK);
}
